// OccupancyEncoder_8237747273947
// MI455X (gfx1250) — matrix operand order repaired and checked by reading, not compiled, not run on hardware
//
#include <hip/hip_runtime.h>
#include <hip/hip_bf16.h>

typedef _Float16 h16;
typedef __attribute__((ext_vector_type(16))) _Float16 v16h;
typedef __attribute__((ext_vector_type(8)))  float    v8f;

#define EPSV 1e-5f

// ---------- WMMA fragment index helpers (16-bit, 16x16x32) ----------
// A-matrix (MxK=16x32): lane l (row=l%16) holds K = {0..7,16..23} (l<16) or {8..15,24..31} (l>=16)
// element e in v16h:  K'(l,e) = (e<8 ? e : e+8) + (l>=16 ? 8 : 0)
// Inverse (place value for row r, K' in [0,32)):
__device__ __forceinline__ void amapInv(int Kp, int& hi, int& e) {
  hi = (Kp >> 3) & 1;
  e  = (Kp & 7) + ((Kp >> 4) & 1) * 8;
}
// B-matrix (KxN=32x16): lane l (col=l%16); same K order as A: K = (e<8 ? e : e+8) + (l>=16 ? 8 : 0)
__device__ __forceinline__ int kmapB(int lane, int e) { return (e < 8 ? e : e + 8) + ((lane & 16) ? 8 : 0); }

// ---------- gfx1250 async global->LDS (ASYNCcnt-tracked, bypasses VGPRs) ----------
__device__ __forceinline__ void asyncLoadB128(unsigned ldsAddr, const void* gptr) {
  asm volatile("global_load_async_to_lds_b128 %0, %1, off"
               :: "v"(ldsAddr), "v"(gptr) : "memory");
}
__device__ __forceinline__ void waitAsync0() {
  asm volatile("s_wait_asynccnt 0" ::: "memory");
}

// ---------- 3x3 SAME conv on one 7x8 channel (fully unrolled, stays in VGPRs) ----------
__device__ __forceinline__ void conv3x3_7x8(const float* __restrict__ X, const float w[9],
                                            float b, float out[56]) {
#pragma unroll
  for (int y = 0; y < 7; ++y) {
#pragma unroll
    for (int xx = 0; xx < 8; ++xx) {
      float a = b;
#pragma unroll
      for (int ky = 0; ky < 3; ++ky) {
        int yy = y + ky - 1;
        if (yy < 0 || yy > 6) continue;
#pragma unroll
        for (int kx = 0; kx < 3; ++kx) {
          int xc = xx + kx - 1;
          if (xc < 0 || xc > 7) continue;
          a += w[ky * 3 + kx] * X[yy * 8 + xc];
        }
      }
      out[y * 8 + xx] = a;
    }
  }
}

// ============================================================================
// Kernel 0: zero the BN-statistics region of the workspace (96 floats)
// ============================================================================
__global__ void k0_zero(float* ws) {
  int t = threadIdx.x;
  if (t < 96) ws[t] = 0.0f;
}

// ============================================================================
// Kernel 1: conv1(+bias) batch statistics (sum, sumsq per channel, 16 ch)
// ============================================================================
__global__ void __launch_bounds__(256) k1_conv1_stats(
    const float* __restrict__ x, const float* __restrict__ w1,
    const float* __restrict__ b1, float* __restrict__ stats) {
  __shared__ float sS[16], sQ[16];
  int tid = threadIdx.x;
  if (tid < 16) { sS[tid] = 0.f; sQ[tid] = 0.f; }
  __syncthreads();

  int sid = blockIdx.x * 256 + tid;
  float X[56];
#pragma unroll
  for (int i = 0; i < 56; i += 4) {
    float4 v = *(const float4*)(x + (size_t)sid * 56 + i);
    X[i] = v.x; X[i + 1] = v.y; X[i + 2] = v.z; X[i + 3] = v.w;
  }
  float out[56];
  for (int o = 0; o < 16; ++o) {
    float w[9];
#pragma unroll
    for (int t = 0; t < 9; ++t) w[t] = w1[o * 9 + t];
    conv3x3_7x8(X, w, b1[o], out);
    float s_ = 0.f, q_ = 0.f;
#pragma unroll
    for (int i = 0; i < 56; ++i) { s_ += out[i]; q_ += out[i] * out[i]; }
    atomicAdd(&sS[o], s_);
    atomicAdd(&sQ[o], q_);
  }
  __syncthreads();
  if (tid < 16) { atomicAdd(&stats[tid], sS[tid]); atomicAdd(&stats[16 + tid], sQ[tid]); }
}

// ============================================================================
// Kernel W: pack fc1_w / (mu_w‖lv_w) into f16 WMMA B-fragment tiles
// ============================================================================
__global__ void __launch_bounds__(256) kW_pack(
    const float* __restrict__ fc1_w, const float* __restrict__ mu_w,
    const float* __restrict__ lv_w, h16* __restrict__ fc1p, h16* __restrict__ headp) {
  int t = blockIdx.x * 256 + threadIdx.x;
  const int NF = 56 * 16 * 512;
  if (t < NF) {
    int tile = t >> 9, w = t & 511;
    int lane = w >> 4, e = w & 15;
    int kt = tile >> 4, nt = tile & 15;
    int K = kt * 32 + kmapB(lane, e);
    int n = nt * 16 + (lane & 15);
    fc1p[t] = (h16)fc1_w[n * 1792 + K];
  } else {
    int t2 = t - NF;
    if (t2 < 8 * 16 * 512) {
      int tile = t2 >> 9, w = t2 & 511;
      int lane = w >> 4, e = w & 15;
      int kt = tile >> 4, nt = tile & 15;
      int K = kt * 32 + kmapB(lane, e);
      int n = nt * 16 + (lane & 15);
      float v = (n < 128) ? mu_w[n * 256 + K] : lv_w[(n - 128) * 256 + K];
      headp[t2] = (h16)v;
    }
  }
}

// ============================================================================
// Kernel 2: recompute conv1, bn1+relu (h1 in LDS), conv2(+bias) -> h2 f16 ws,
//           bn2 stats into ws[32..95]. 256 threads, 16 samples / block.
// ============================================================================
__global__ void __launch_bounds__(256) k2_conv2(
    const float* __restrict__ x, const float* __restrict__ w1,
    const float* __restrict__ b1, const float* __restrict__ bn1w,
    const float* __restrict__ bn1b, const float* __restrict__ w2,
    const float* __restrict__ b2, const float* __restrict__ stats1,
    h16* __restrict__ h2, float* __restrict__ stats2, float N1f) {
  __shared__ h16 h1L[16 * 16 * 56];   // 28KB
  __shared__ float w2L[32 * 16 * 9];  // 18KB
  __shared__ float sS[32], sQ[32];

  int tid = threadIdx.x;
  int b0 = blockIdx.x * 16;
  for (int i = tid; i < 4608; i += 256) w2L[i] = w2[i];
  if (tid < 32) { sS[tid] = 0.f; sQ[tid] = 0.f; }

  // ---- stage 1: h1 = relu(bn1(conv1(x))), one thread = (sample, cin)
  {
    int s = tid >> 4, ch = tid & 15;
    float X[56];
#pragma unroll
    for (int i = 0; i < 56; i += 4) {
      float4 v = *(const float4*)(x + (size_t)(b0 + s) * 56 + i);
      X[i] = v.x; X[i + 1] = v.y; X[i + 2] = v.z; X[i + 3] = v.w;
    }
    float mean = stats1[ch] / N1f;
    float var  = stats1[16 + ch] / N1f - mean * mean;
    float sc   = bn1w[ch] * rsqrtf(var + EPSV);
    float sh   = bn1b[ch] - mean * sc;
    float w[9];
#pragma unroll
    for (int t = 0; t < 9; ++t) w[t] = w1[ch * 9 + t];
    float out[56];
    conv3x3_7x8(X, w, b1[ch], out);
#pragma unroll
    for (int i = 0; i < 56; ++i)
      h1L[(s * 16 + ch) * 56 + i] = (h16)fmaxf(0.f, out[i] * sc + sh);
  }
  __syncthreads();

  // ---- stage 2: conv2, tasks = (s, cout, y) -> 8 outputs each
  for (int task = tid; task < 16 * 32 * 7; task += 256) {
    int s = task / 224;
    int r = task % 224;
    int cout = r / 7, y = r % 7;
    float acc[8];
    float bb = b2[cout];
#pragma unroll
    for (int xx = 0; xx < 8; ++xx) acc[xx] = bb;
    for (int cin = 0; cin < 16; ++cin) {
#pragma unroll
      for (int ky = 0; ky < 3; ++ky) {
        int yy = y + ky - 1;
        if (yy < 0 || yy > 6) continue;
        const h16* row = &h1L[(s * 16 + cin) * 56 + yy * 8];
        float rr[8];
#pragma unroll
        for (int xx = 0; xx < 8; ++xx) rr[xx] = (float)row[xx];
        float wa = w2L[((cout * 16 + cin) * 3 + ky) * 3 + 0];
        float wb = w2L[((cout * 16 + cin) * 3 + ky) * 3 + 1];
        float wc = w2L[((cout * 16 + cin) * 3 + ky) * 3 + 2];
#pragma unroll
        for (int xx = 0; xx < 8; ++xx) {
          if (xx > 0) acc[xx] += wa * rr[xx - 1];
          acc[xx] += wb * rr[xx];
          if (xx < 7) acc[xx] += wc * rr[xx + 1];
        }
      }
    }
    float ls = 0.f, lq = 0.f;
    union { h16 h[8]; uint4 u; } pk;
#pragma unroll
    for (int xx = 0; xx < 8; ++xx) {
      ls += acc[xx]; lq += acc[xx] * acc[xx];
      pk.h[xx] = (h16)acc[xx];
    }
    size_t off = ((size_t)(b0 + s) * 32 + cout) * 56 + y * 8;
    *((uint4*)(h2 + off)) = pk.u;
    atomicAdd(&sS[cout], ls);
    atomicAdd(&sQ[cout], lq);
  }
  __syncthreads();
  if (tid < 32) { atomicAdd(&stats2[tid], sS[tid]); atomicAdd(&stats2[32 + tid], sQ[tid]); }
}

// ============================================================================
// Kernel 3 (fused): async-LDS stage-in -> bn2+relu -> attention -> fc1 (WMMA)
//                   -> head (WMMA) -> mu/logvar.  512 threads = 16 waves.
// ============================================================================
__global__ void __launch_bounds__(512) k3_main(
    const h16* __restrict__ h2, const float* __restrict__ ws,
    const float* __restrict__ bn2w, const float* __restrict__ bn2b,
    const float* __restrict__ q_w, const float* __restrict__ q_b,
    const float* __restrict__ k_w, const float* __restrict__ k_b,
    const float* __restrict__ v_w, const float* __restrict__ v_b,
    const float* __restrict__ gammaPtr,
    const h16* __restrict__ fc1p, const float* __restrict__ fc1_b,
    const h16* __restrict__ headp, const float* __restrict__ mu_b,
    const float* __restrict__ lv_b, float* __restrict__ dout,
    int NB, float N2f) {
  __shared__ alignas(32) h16 Hbuf[16 * 32 * 56];  // 57344 B  post-bn2 activations
  __shared__ alignas(32) h16 Afrag[56 * 512];     // 57344 B  fc1 A fragments
  __shared__ alignas(32) h16 F2frag[8 * 512];     // 8192 B   head A fragments
  __shared__ float qkbuf[16 * 8 * 56];            // 28672 B  q(4)+k(4) per sample
  __shared__ float vwL[32 * 32];
  __shared__ float vbL[32], qwL[128], kwL[128], qbL[4], kbL[4];
  __shared__ float sc2[32], sh2[32];

  int tid = threadIdx.x;
  int lane = tid & 31;
  int wave = tid >> 5;
  int b0 = blockIdx.x * 16;

  // ---- stage A-async: kick off the 57KB h2 tile -> LDS via ASYNCcnt DMA path
  {
    const char* gsrc = (const char*)(h2 + (size_t)b0 * 1792);
    unsigned ldsBase = (unsigned)(uintptr_t)(&Hbuf[0]);
#pragma unroll
    for (int v = 0; v < 7; ++v) {
      int i16 = tid + v * 512;  // 3584 x 16B chunks
      asyncLoadB128(ldsBase + (unsigned)i16 * 16u, gsrc + (size_t)i16 * 16);
    }
  }

  // ---- stage A0 (overlapped with async DMA): params + bn2 scale/shift
  if (tid < 32) {
    float sum = ws[32 + tid], sq = ws[64 + tid];
    float mean = sum / N2f, var = sq / N2f - mean * mean;
    float sc = bn2w[tid] * rsqrtf(var + EPSV);
    sc2[tid] = sc;
    sh2[tid] = bn2b[tid] - mean * sc;
    vbL[tid] = v_b[tid];
  }
  for (int i = tid; i < 1024; i += 512) vwL[i] = v_w[i];
  if (tid < 128) { qwL[tid] = q_w[tid]; kwL[tid] = k_w[tid]; }
  if (tid < 4) { qbL[tid] = q_b[tid]; kbL[tid] = k_b[tid]; }
  waitAsync0();
  __syncthreads();

  // ---- stage A1: in-place bn2+relu on the async-staged tile
  for (int idx = tid; idx < 16 * 32 * 56; idx += 512) {
    int c = (idx / 56) & 31;
    float v = (float)Hbuf[idx];
    Hbuf[idx] = (h16)fmaxf(0.f, v * sc2[c] + sh2[c]);
  }
  __syncthreads();

  // ---- stage C1: q,k projections per (sample, position)
  for (int task = tid; task < 16 * 56; task += 512) {
    int s = task / 56, n = task % 56;
    float qa[4], ka[4];
#pragma unroll
    for (int o = 0; o < 4; ++o) { qa[o] = qbL[o]; ka[o] = kbL[o]; }
    for (int c = 0; c < 32; ++c) {
      float hv = (float)Hbuf[(s * 32 + c) * 56 + n];
#pragma unroll
      for (int o = 0; o < 4; ++o) {
        qa[o] += qwL[o * 32 + c] * hv;
        ka[o] += kwL[o * 32 + c] * hv;
      }
    }
#pragma unroll
    for (int o = 0; o < 4; ++o) {
      qkbuf[(s * 8 + o) * 56 + n] = qa[o];
      qkbuf[(s * 8 + 4 + o) * 56 + n] = ka[o];
    }
  }
  __syncthreads();

  // ---- stage C2: attention row per lane; write residual result into Afrag
  {
    int s = wave;  // one sample per wave
    float gam = gammaPtr[0];
    for (int pass = 0; pass < 2; ++pass) {
      int i = pass * 32 + lane;
      if (i < 56) {
        float qi[4];
#pragma unroll
        for (int o = 0; o < 4; ++o) qi[o] = qkbuf[(s * 8 + o) * 56 + i];
        float p[56];
#pragma unroll
        for (int j = 0; j < 56; ++j) {
          p[j] = qi[0] * qkbuf[(s * 8 + 4) * 56 + j]
               + qi[1] * qkbuf[(s * 8 + 5) * 56 + j]
               + qi[2] * qkbuf[(s * 8 + 6) * 56 + j]
               + qi[3] * qkbuf[(s * 8 + 7) * 56 + j];
        }
        float m = -1e30f;
#pragma unroll
        for (int j = 0; j < 56; ++j) m = fmaxf(m, p[j]);
        float ssum = 0.f;
#pragma unroll
        for (int j = 0; j < 56; ++j) { p[j] = __expf(p[j] - m); ssum += p[j]; }
        float inv = 1.0f / ssum;
#pragma unroll
        for (int j = 0; j < 56; ++j) p[j] *= inv;
        // t[c'] = sum_j p[j] * h[c', j]  (softmax rows sum to 1 -> v_b passes through)
        float t[32];
#pragma unroll
        for (int c2 = 0; c2 < 32; ++c2) t[c2] = 0.f;
#pragma unroll
        for (int j = 0; j < 56; ++j) {
          float pj = p[j];
#pragma unroll
          for (int c2 = 0; c2 < 32; ++c2)
            t[c2] += pj * (float)Hbuf[(s * 32 + c2) * 56 + j];
        }
#pragma unroll
        for (int c = 0; c < 32; ++c) {
          float o_ = vbL[c];
#pragma unroll
          for (int c2 = 0; c2 < 32; ++c2) o_ += vwL[c * 32 + c2] * t[c2];
          float res = (float)Hbuf[(s * 32 + c) * 56 + i];
          float val = gam * o_ + res;
          int kk = c * 56 + i;          // f index (C-major flatten)
          int kt = kk >> 5, Kp = kk & 31, hi, e;
          amapInv(Kp, hi, e);
          Afrag[kt * 512 + (s + hi * 16) * 16 + e] = (h16)val;
        }
      }
    }
  }
  __syncthreads();

  // ---- stage D: fc1 GEMM  D(16x256) = A(16x1792) x B(1792x256), wave = n-tile
  {
    int nt = wave;
    v8f acc = {};
#pragma unroll 4
    for (int kt = 0; kt < 56; ++kt) {
      v16h a = *(const v16h*)(&Afrag[kt * 512 + lane * 16]);
      v16h b = *(const v16h*)(&fc1p[(size_t)(kt * 16 + nt) * 512 + lane * 16]);
      acc = __builtin_amdgcn_wmma_f32_16x16x32_f16(false, a, false, b, (short)0, acc,
                                                   false, false);
    }
    int n = nt * 16 + (lane & 15);
    float bias = fc1_b[n];
    int rbase = (lane & 16) ? 8 : 0;
    int kt2 = n >> 5, Kp = n & 31, hi, e;
    amapInv(Kp, hi, e);
#pragma unroll
    for (int r = 0; r < 8; ++r) {
      int srow = r + rbase;
      float val = fmaxf(0.f, acc[r] + bias);
      F2frag[kt2 * 512 + (srow + hi * 16) * 16 + e] = (h16)val;
    }
  }
  __syncthreads();

  // ---- stage E: head GEMM  (16x256) = f2(16x256) x [mu_w;lv_w]^T, + bias, store
  {
    int nt = wave;
    v8f acc = {};
#pragma unroll
    for (int kt = 0; kt < 8; ++kt) {
      v16h a = *(const v16h*)(&F2frag[kt * 512 + lane * 16]);
      v16h b = *(const v16h*)(&headp[(size_t)(kt * 16 + nt) * 512 + lane * 16]);
      acc = __builtin_amdgcn_wmma_f32_16x16x32_f16(false, a, false, b, (short)0, acc,
                                                   false, false);
    }
    int ng = nt * 16 + (lane & 15);
    int rbase = (lane & 16) ? 8 : 0;
    float bias;
    float* outCol;
    if (ng < 128) {
      bias = mu_b[ng];
      outCol = dout + ng;
    } else {
      bias = lv_b[ng - 128];
      outCol = dout + (size_t)NB * 128 + (ng - 128);
    }
#pragma unroll
    for (int r = 0; r < 8; ++r) {
      int srow = r + rbase;
      outCol[(size_t)(b0 + srow) * 128] = acc[r] + bias;
    }
  }
}

// ============================================================================
extern "C" void kernel_launch(void* const* d_in, const int* in_sizes, int n_in,
                              void* d_out, int out_size, void* d_ws, size_t ws_size,
                              hipStream_t stream) {
  const float* x       = (const float*)d_in[0];
  const float* conv1_w = (const float*)d_in[1];
  const float* conv1_b = (const float*)d_in[2];
  const float* bn1_w   = (const float*)d_in[3];
  const float* bn1_b   = (const float*)d_in[4];
  const float* conv2_w = (const float*)d_in[5];
  const float* conv2_b = (const float*)d_in[6];
  const float* bn2_w   = (const float*)d_in[7];
  const float* bn2_b   = (const float*)d_in[8];
  const float* q_w     = (const float*)d_in[9];
  const float* q_b     = (const float*)d_in[10];
  const float* k_w     = (const float*)d_in[11];
  const float* k_b     = (const float*)d_in[12];
  const float* v_w     = (const float*)d_in[13];
  const float* v_b     = (const float*)d_in[14];
  const float* gamma   = (const float*)d_in[15];
  const float* fc1_w   = (const float*)d_in[16];
  const float* fc1_b   = (const float*)d_in[17];
  const float* mu_w    = (const float*)d_in[18];
  const float* mu_b    = (const float*)d_in[19];
  const float* lv_w    = (const float*)d_in[20];
  const float* lv_b    = (const float*)d_in[21];

  int NB = in_sizes[0] / 56;           // 32768
  float N1f = (float)NB * 56.0f;

  float* ws = (float*)d_ws;
  // ws layout (floats): [0..31] bn1 stats, [32..95] bn2 stats,
  // [128..) fc1 packed f16 (458752 h16 = 229376 f, in a 114688 f slot), then head packed f16
  // (65536 h16 = 32768 f, in a 16384 f slot), then h2 f16 (NB*1792 h16).
  h16* fc1p  = (h16*)(ws + 128);
  h16* headp = (h16*)(ws + 128 + 114688);
  h16* h2    = (h16*)(ws + 128 + 114688 + 16384);
  float* dout = (float*)d_out;

  k0_zero<<<1, 128, 0, stream>>>(ws);
  k1_conv1_stats<<<NB / 256, 256, 0, stream>>>(x, conv1_w, conv1_b, ws);
  kW_pack<<<2048, 256, 0, stream>>>(fc1_w, mu_w, lv_w, fc1p, headp);
  k2_conv2<<<NB / 16, 256, 0, stream>>>(x, conv1_w, conv1_b, bn1_w, bn1_b,
                                        conv2_w, conv2_b, ws, h2, ws + 32, N1f);
  k3_main<<<NB / 16, 512, 0, stream>>>(h2, ws, bn2_w, bn2_b, q_w, q_b, k_w, k_b,
                                       v_w, v_b, gamma, fc1p, fc1_b, headp,
                                       mu_b, lv_b, dout, NB, N1f);
}
